// TritonGatherConv_35734127903504
// MI455X (gfx1250) — matrix operand order repaired and checked by reading, not compiled, not run on hardware
//
#include <hip/hip_runtime.h>
#include <hip/hip_bf16.h>

// ---------------------------------------------------------------------------
// Types for CDNA5 WMMA (wave32): V_WMMA_F32_16X16X32_BF16
// ---------------------------------------------------------------------------
typedef __attribute__((ext_vector_type(16))) __bf16 v16bf;
typedef __attribute__((ext_vector_type(8)))  float  v8f;

union FragBF16 {
    unsigned u[8];   // 8 VGPRs, 2 packed bf16 each
    v16bf    v;
};

__device__ __forceinline__ unsigned pack2_bf16(float lo, float hi) {
    // round-to-nearest-even f32 -> bf16, packed into one dword
    unsigned a = __builtin_bit_cast(unsigned, lo);
    unsigned b = __builtin_bit_cast(unsigned, hi);
    a = (a + 0x7FFFu + ((a >> 16) & 1u)) >> 16;
    b = (b + 0x7FFFu + ((b >> 16) & 1u)) >> 16;
    return (b << 16) | (a & 0xFFFFu);
}

__device__ __forceinline__ unsigned short bf16_1(float f) {
    unsigned a = __builtin_bit_cast(unsigned, f);
    return (unsigned short)((a + 0x7FFFu + ((a >> 16) & 1u)) >> 16);
}

__device__ __forceinline__ float fast_sigmoid(float z) {
    return __builtin_amdgcn_rcpf(1.0f + __expf(-z));     // v_rcp_f32, no div chain
}
__device__ __forceinline__ float silu_f(float v) {
    return v * fast_sigmoid(v);
}
__device__ __forceinline__ float fast_tanh(float z) {
    // tanh(z) = 1 - 2/(exp(2z)+1)
    return 1.0f - 2.0f * __builtin_amdgcn_rcpf(__expf(2.0f * z) + 1.0f);
}

// ---------------------------------------------------------------------------
// f32 -> packed bf16 converter (memory-bound; n2 = n/2 dword-pairs)
// ---------------------------------------------------------------------------
__global__ __launch_bounds__(256)
void cvt_f32_bf16(const float* __restrict__ src, unsigned* __restrict__ dst, int n2) {
    int i = blockIdx.x * 256 + threadIdx.x;
    if (i < n2) dst[i] = pack2_bf16(src[2 * i], src[2 * i + 1]);
}

// ---------------------------------------------------------------------------
// GEMM: Y[M,N] = silu(X[M,Kd] @ W[N,Kd]^T + bias[N]), X/W pre-packed bf16.
// block = 256 threads = 8 waves; block tile 16(M) x 128(N); K unrolled x2
// with double-buffered A tile in LDS and two accumulator chains.
// ---------------------------------------------------------------------------
__global__ __launch_bounds__(256)
void wmma_gemm_bias_silu(const __bf16* __restrict__ X,
                         const __bf16* __restrict__ W,
                         const float* __restrict__ bias,
                         float* __restrict__ Y,
                         int M, int N, int Kd) {
    __shared__ unsigned alds[2][16][17];   // two 16x32-bf16 tiles (dword packed)

    const int tid  = threadIdx.x;
    const int wave = tid >> 5;
    const int lane = tid & 31;
    const int lm   = lane & 15;            // N-column (B/D) or M-row (A) in tile
    const int hi   = lane >> 4;            // lane half selects K-half

    const int m0 = blockIdx.y * 16;
    const int n0 = blockIdx.x * 128 + wave * 16;
    const bool active = (n0 < N);

    const int aRow = tid >> 4;             // 0..15
    const int aCol = tid & 15;             // dword column within 32-wide tile

    v8f acc0 = {0.f,0.f,0.f,0.f,0.f,0.f,0.f,0.f};
    v8f acc1 = {0.f,0.f,0.f,0.f,0.f,0.f,0.f,0.f};

    const unsigned* xrow = (const unsigned*)(X + (size_t)(m0 + aRow) * Kd);
    const unsigned* wrow = active ? (const unsigned*)(W + (size_t)(n0 + lm) * Kd) : nullptr;

    for (int k0 = 0; k0 < Kd; k0 += 64) {
        // ---- stage two 16x32 bf16 A tiles (k0, k0+32) into LDS ----
        const unsigned* xp = xrow + (k0 >> 1);
        alds[0][aRow][aCol] = xp[aCol];
        alds[1][aRow][aCol] = xp[16 + aCol];
        __syncthreads();

        if (active) {
            // B fragments: column n = n0+lm; K rows (k0|k0+32) + hi*16 + 0..15
            const unsigned* wp = wrow + (k0 >> 1) + hi * 8;
            FragBF16 b0, b1, a0, a1;
            #pragma unroll
            for (int j = 0; j < 8; ++j) { b0.u[j] = wp[j]; b1.u[j] = wp[16 + j]; }

            const int cb = hi * 8;
            #pragma unroll
            for (int j = 0; j < 4; ++j) {
                a0.u[j]     = alds[0][lm][cb + j];
                a0.u[4 + j] = alds[0][lm][cb + 4 + j];
                a1.u[j]     = alds[1][lm][cb + j];
                a1.u[4 + j] = alds[1][lm][cb + 4 + j];
            }
            acc0 = __builtin_amdgcn_wmma_f32_16x16x32_bf16(
                       false, a0.v, false, b0.v, (short)0, acc0, false, false);
            acc1 = __builtin_amdgcn_wmma_f32_16x16x32_bf16(
                       false, a1.v, false, b1.v, (short)0, acc1, false, false);
        }
        __syncthreads();
    }

    if (active) {
        const v8f acc = acc0 + acc1;
        const int n = n0 + lm;
        const float bv = (bias != nullptr) ? bias[n] : 0.0f;
        #pragma unroll
        for (int r = 0; r < 8; ++r) {
            const int m = m0 + r + hi * 8;     // C/D layout: lanes>=16 hold M=r+8
            if (m < M)
                Y[(size_t)m * N + n] = silu_f(acc[r] + bv);
        }
    }
}

// ---------------------------------------------------------------------------
// Gather-conv. Prologue (132 threads): per (head, sample) compute interpolated
// kernel weight + gather row offset into LDS. Main loop: 4 heads x 64 dims,
// uniform LDS reads + coalesced x gather + FMA. Writes hidden as bf16.
// ---------------------------------------------------------------------------
#define GC_MAXFREQ 16.0f
#define GC_MINFREQ 1.0f

__global__ __launch_bounds__(256)
void gather_conv_kernel(const float* __restrict__ x,
                        const float* __restrict__ waveY,       // [B*L, 32]
                        const float* __restrict__ kernY,       // [B*L, 1024]
                        unsigned short* __restrict__ hidden,   // [B*L, 1024] bf16
                        int L) {
    __shared__ float w_lds[4][33];
    __shared__ int   p_lds[4][33];

    const int bl  = blockIdx.x;                 // 0 .. B*L-1
    const int h0  = blockIdx.y * 4;
    const int tid = threadIdx.x;
    const int l   = bl % L;
    const int rowbase = bl - l;                 // = b*L

    if (tid < 132) {
        const int hl = tid / 33;
        const int s  = tid - hl * 33;
        const int h  = h0 + hl;

        const float wv0   = waveY[bl * 32 + h];
        const float wv1   = waveY[bl * 32 + 16 + h];
        const float freq  = (GC_MAXFREQ - GC_MINFREQ) * fast_sigmoid(wv0) + GC_MINFREQ;
        const float phase = fast_tanh(wv1) * GC_MAXFREQ;

        const float off   = (float)(s - 16) * freq;
        const float pos_i = rintf((float)l + off + phase);     // round-half-even
        const float lmax  = (float)(L - 1);
        const bool  valid = (pos_i >= 0.0f) && (pos_i <= lmax);
        const int   pc    = (int)fminf(fmaxf(pos_i, 0.0f), lmax);

        const float kp   = (off + 256.0f) * (63.0f / 512.0f);
        const float k0f  = floorf(kp);
        const float frac = kp - k0f;
        const int   k0   = min(max((int)k0f, 0), 63);
        const int   k1   = min(k0 + 1, 63);

        const float* kb = kernY + bl * 1024 + h * 64;
        w_lds[hl][s] = valid ? (kb[k0] * (1.0f - frac) + kb[k1] * frac) : 0.0f;
        p_lds[hl][s] = (rowbase + pc) * 1024 + h * 64;
    }
    __syncthreads();

    const int hl = tid >> 6;
    const int d  = tid & 63;
    float acc = 0.0f;
    #pragma unroll 3
    for (int s = 0; s < 33; ++s)
        acc = fmaf(w_lds[hl][s], x[p_lds[hl][s] + d], acc);

    hidden[bl * 1024 + (h0 + hl) * 64 + d] = bf16_1(acc);
}

// ---------------------------------------------------------------------------
// Launch: bf16 conversions -> wave GEMM -> kern GEMM -> gather -> out GEMM
// ---------------------------------------------------------------------------
extern "C" void kernel_launch(void* const* d_in, const int* in_sizes, int n_in,
                              void* d_out, int out_size, void* d_ws, size_t ws_size,
                              hipStream_t stream) {
    const float* x        = (const float*)d_in[0];
    const float* wave_w   = (const float*)d_in[1];
    const float* wave_b   = (const float*)d_in[2];
    const float* kernel_w = (const float*)d_in[3];
    const float* kernel_b = (const float*)d_in[4];
    const float* out_w    = (const float*)d_in[5];
    float* out = (float*)d_out;

    const int B = 2, L = 2048, C = 1024;
    const int M = B * L;                         // 4096 rows

    // workspace carve-up (256B aligned)
    char* ws = (char*)d_ws;
    size_t off = 0;
    auto carve = [&](size_t bytes) {
        void* p = ws + off;
        off += (bytes + 255) & ~(size_t)255;
        return p;
    };
    __bf16* x_bf    = (__bf16*)carve((size_t)M * C * 2);        // 8 MB
    __bf16* ww_bf   = (__bf16*)carve((size_t)32 * C * 2);
    __bf16* kw_bf   = (__bf16*)carve((size_t)C * C * 2);        // 2 MB
    __bf16* ow_bf   = (__bf16*)carve((size_t)C * C * 2);        // 2 MB
    float*  waveY   = (float*) carve((size_t)M * 32 * 4);
    float*  kernY   = (float*) carve((size_t)M * C * 4);        // 16 MB
    unsigned short* hidden_bf = (unsigned short*)carve((size_t)M * C * 2); // 8 MB

    const dim3 blk(256);

    // 0) one-shot bf16 conversions (memory-bound, halves GEMM read traffic)
    {
        int n2;
        n2 = M * C / 2;  cvt_f32_bf16<<<(n2 + 255) / 256, blk, 0, stream>>>(x,        (unsigned*)x_bf,  n2);
        n2 = 32 * C / 2; cvt_f32_bf16<<<(n2 + 255) / 256, blk, 0, stream>>>(wave_w,   (unsigned*)ww_bf, n2);
        n2 = C * C / 2;  cvt_f32_bf16<<<(n2 + 255) / 256, blk, 0, stream>>>(kernel_w, (unsigned*)kw_bf, n2);
        n2 = C * C / 2;  cvt_f32_bf16<<<(n2 + 255) / 256, blk, 0, stream>>>(out_w,    (unsigned*)ow_bf, n2);
    }

    // 1) wave projection: N = 32
    wmma_gemm_bias_silu<<<dim3(1, M / 16), blk, 0, stream>>>(
        x_bf, ww_bf, wave_b, waveY, M, 32, C);

    // 2) dynamic kernel projection: N = 1024
    wmma_gemm_bias_silu<<<dim3(C / 128, M / 16), blk, 0, stream>>>(
        x_bf, kw_bf, kernel_b, kernY, M, C, C);

    // 3) data-dependent gather-conv (writes hidden as bf16)
    gather_conv_kernel<<<dim3(M, 4), blk, 0, stream>>>(
        x, waveY, kernY, hidden_bf, L);

    // 4) output projection + silu -> f32 result
    wmma_gemm_bias_silu<<<dim3(C / 128, M / 16), blk, 0, stream>>>(
        (const __bf16*)hidden_bf, ow_bf, nullptr, out, M, C, C);
}
